// Attention_18786186953586
// MI455X (gfx1250) — matrix operand order repaired and checked by reading, not compiled, not run on hardware
//
#include <hip/hip_runtime.h>
#include <hip/hip_bf16.h>

typedef unsigned short u16;
typedef unsigned int   u32;
typedef unsigned long long u64;

typedef __attribute__((ext_vector_type(16))) __bf16 v16bf;
typedef __attribute__((ext_vector_type(8)))  float  v8f;
typedef __attribute__((ext_vector_type(4)))  unsigned int u32x4;
typedef __attribute__((ext_vector_type(8)))  int i32x8;
typedef __attribute__((ext_vector_type(4)))  int i32x4;

union FragB { u32 u[8];  v16bf v; };
union FragF { float f[8]; v8f  v; };

#define B_  2
#define T_  4096
#define D_  896
#define H_  7
#define HD_ 128

#if __has_builtin(__builtin_amdgcn_tensor_load_to_lds) && __has_builtin(__builtin_amdgcn_s_wait_tensorcnt)
#define USE_TDM 1
#else
#define USE_TDM 0
#endif

__device__ inline u16 f2bf(float f) {
  union { __bf16 h; u16 u; } cv;
  cv.h = (__bf16)f;                 // native v_cvt, RNE
  return cv.u;
}

__device__ inline v8f wmma_bf16(v16bf a, v16bf b, v8f c) {
  return __builtin_amdgcn_wmma_f32_16x16x32_bf16(false, a, false, b, (short)0, c, false, false);
}

// A-fragment K offset for dword j (16-bit A 16x32 layout, ISA 7.12.2)
__device__ inline int a_k0(int kg, int j) { return kg + 2 * (j & 3) + ((j >> 2) * 16); }

#if USE_TDM
// Generic TDM 2D tile load: tile_w dwords x tile_h rows from a row-major
// tensor (row stride = stride_dw dwords) into LDS, with 1 padding dword
// inserted every (1<<(pad_code+1)) dwords. D# packing per
// CDNA5 ISA (data_size=4B, type=2, count=1).
__device__ inline void tdm_load_2d(const void* gsrc, u32 lds_addr,
                                   u32 tile_w, u32 tile_h, u32 stride_dw, u32 pad_code) {
  const u64 ga = (u64)(size_t)gsrc;
  u32x4 g0;
  g0.x = 1u;                                             // count=1, user mode
  g0.y = lds_addr;                                       // lds_addr (bytes)
  g0.z = (u32)ga;                                        // global_addr[31:0]
  g0.w = ((u32)(ga >> 32) & 0x01FFFFFFu) | (2u << 30);   // addr[56:32] | type=2
  i32x8 g1;
  g1[0] = (int)((2u << 16) | (1u << 20) | (pad_code << 22)); // 4B units, pad_en, interval
  g1[1] = (int)(tile_w << 16);                           // tensor_dim0.lo16
  g1[2] = (int)((tile_h & 0xffffu) << 16);               // dim0.hi=0 | tensor_dim1.lo16
  g1[3] = (int)(((tile_h >> 16) & 0xffffu) | (tile_w << 16)); // dim1.hi | tile_dim0
  g1[4] = (int)tile_h;                                   // tile_dim1 | tile_dim2=0
  g1[5] = (int)stride_dw;                                // tensor_dim0_stride.lo32
  g1[6] = 0;
  g1[7] = 0;
  i32x4 z4 = {0, 0, 0, 0};
#if defined(__clang_major__) && (__clang_major__ >= 23)
  i32x8 z8 = {0, 0, 0, 0, 0, 0, 0, 0};
  __builtin_amdgcn_tensor_load_to_lds(g0, g1, z4, z4, z8, 0);
#else
  __builtin_amdgcn_tensor_load_to_lds(g0, g1, z4, z4, 0);
#endif
}
#endif

// ---------------------------------------------------------------------------
// GEMM: C = A(MxK) @ W(KxN). 128x64 tile per block (4 waves x 32x64 strips),
// bf16 WMMA f32 accumulate, double-buffered LDS staging + HBM prefetch.
// AF32: A is fp32 (convert while staging); else raw bf16 (TDM-staged).
// EP:   0 = fp32 store, 1 = bf16 store, 2 = fused interleaved-RoPE bf16 store
// ---------------------------------------------------------------------------
template<bool AF32, int EP>
__global__ __launch_bounds__(128) void gemm_wmma_kernel(
    const void* __restrict__ Ap, const float* __restrict__ W, void* __restrict__ outp,
    int M, int N, int K, const float* __restrict__ cosT, const float* __restrict__ sinT)
{
  __shared__ u16 As[2][128 * 34];                    // [row][k] pad 34 (17 dw/row)
  __shared__ u16 Ws[2][64 * 34];                     // transposed [n][k] pad 34
  __shared__ float Cs[(EP == 2) ? 128 * 66 : 1];     // rope epilogue bounce

  const int tid  = threadIdx.x;
  const int wid  = tid >> 5;
  const int lane = tid & 31;
  const int l16  = lane & 15;
  const int lh   = lane >> 4;
  const int kg   = lh * 8;
  const int moff = lh * 8;
  const int rowBase = blockIdx.y * 128;
  const int nBase   = blockIdx.x * 64;

  FragF acc[2][4];
#pragma unroll
  for (int mt = 0; mt < 2; ++mt)
#pragma unroll
    for (int nt = 0; nt < 4; ++nt)
#pragma unroll
      for (int r = 0; r < 8; ++r) acc[mt][nt].f[r] = 0.f;

  auto stage = [&](int kk, int buf) {
    if constexpr (AF32) {
      const float* A = (const float*)Ap;
#pragma unroll 4
      for (int i = 0; i < 32; ++i) {
        int e = tid + i * 128;
        int r = e >> 5, c = e & 31;
        As[buf][r * 34 + c] = f2bf(A[(size_t)(rowBase + r) * K + kk + c]);
      }
      if (kk + 64 < K)   // prefetch panel after next (one cacheline per thread)
        __builtin_prefetch(&A[(size_t)(rowBase + tid) * K + kk + 64], 0, 0);
    } else {
      const u16* A = (const u16*)Ap;
#if USE_TDM
      if (tid < 32)      // 128 rows x 16 dw, row stride K/2 dw, +1 dw pad per 16 dw
        tdm_load_2d(&A[(size_t)rowBase * K + kk], (u32)(size_t)&As[buf][0],
                    16u, 128u, (u32)(K >> 1), 3u);
#else
#pragma unroll 4
      for (int i = 0; i < 32; ++i) {
        int e = tid + i * 128;
        int r = e >> 5, c = e & 31;
        As[buf][r * 34 + c] = A[(size_t)(rowBase + r) * K + kk + c];
      }
      if (kk + 64 < K)
        __builtin_prefetch(&A[(size_t)(rowBase + tid) * K + kk + 64], 0, 0);
#endif
    }
#pragma unroll 4
    for (int i = 0; i < 16; ++i) {
      int e = tid + i * 128;
      int c = e & 63, r = e >> 6;
      Ws[buf][c * 34 + r] = f2bf(W[(size_t)(kk + r) * N + nBase + c]);
    }
    if (tid < 64 && kk + 64 < K)
      __builtin_prefetch(&W[(size_t)(kk + 64 + (tid & 31)) * N + nBase + (tid >> 5) * 32], 0, 0);
  };

  stage(0, 0);
  int cur = 0;
  for (int kk = 0; kk < K; kk += 32) {
#if USE_TDM
    if constexpr (!AF32) {
      if (tid < 32) __builtin_amdgcn_s_wait_tensorcnt(0);  // A DMA done before barrier
    }
#endif
    __syncthreads();
    if (kk + 32 < K) stage(kk + 32, cur ^ 1);   // overlap next tile with compute

    FragB a[2];
#pragma unroll
    for (int mt = 0; mt < 2; ++mt) {
      const int m = wid * 32 + mt * 16 + l16;
#pragma unroll
      for (int j = 0; j < 8; ++j)
        a[mt].u[j] = *(const u32*)&As[cur][m * 34 + a_k0(kg, j)];
    }
    auto loadBW = [&](int nt, FragB& bw) {
      const int n = nt * 16 + l16;
#pragma unroll
      for (int j = 0; j < 8; ++j)
        bw.u[j] = *(const u32*)&Ws[cur][n * 34 + a_k0(kg, j)];
    };
    FragB bw[2];
    loadBW(0, bw[0]);
#pragma unroll
    for (int nt = 0; nt < 4; ++nt) {            // 1-deep pipelined B fragments
      if (nt + 1 < 4) loadBW(nt + 1, bw[(nt + 1) & 1]);
      acc[0][nt].v = wmma_bf16(a[0].v, bw[nt & 1].v, acc[0][nt].v);
      acc[1][nt].v = wmma_bf16(a[1].v, bw[nt & 1].v, acc[1][nt].v);
    }
    cur ^= 1;
  }

  if constexpr (EP == 0) {
    float* out = (float*)outp;
#pragma unroll
    for (int mt = 0; mt < 2; ++mt)
#pragma unroll
      for (int nt = 0; nt < 4; ++nt)
#pragma unroll
        for (int r = 0; r < 8; ++r)
          out[(size_t)(rowBase + wid * 32 + mt * 16 + r + moff) * N + nBase + nt * 16 + l16] =
              acc[mt][nt].f[r];
  } else if constexpr (EP == 1) {
    u16* out = (u16*)outp;
#pragma unroll
    for (int mt = 0; mt < 2; ++mt)
#pragma unroll
      for (int nt = 0; nt < 4; ++nt)
#pragma unroll
        for (int r = 0; r < 8; ++r)
          out[(size_t)(rowBase + wid * 32 + mt * 16 + r + moff) * N + nBase + nt * 16 + l16] =
              f2bf(acc[mt][nt].f[r]);
  } else {
    __syncthreads();
#pragma unroll
    for (int mt = 0; mt < 2; ++mt)
#pragma unroll
      for (int nt = 0; nt < 4; ++nt)
#pragma unroll
        for (int r = 0; r < 8; ++r)
          Cs[(wid * 32 + mt * 16 + r + moff) * 66 + nt * 16 + l16] = acc[mt][nt].f[r];
    __syncthreads();
    u16* out = (u16*)outp;
#pragma unroll 4
    for (int i = 0; i < 32; ++i) {
      int e = tid + i * 128;
      int r = e >> 5, p = e & 31;
      int nl = 2 * p;
      float x0 = Cs[r * 66 + nl], x1 = Cs[r * 66 + nl + 1];
      int gn   = nBase + nl;
      int grow = rowBase + r;
      int t    = grow % T_;
      int pj   = (gn & (HD_ - 1)) >> 1;
      float cv = cosT[t * (HD_ / 2) + pj], sv = sinT[t * (HD_ / 2) + pj];
      float o0 = x0 * cv - x1 * sv;
      float o1 = x0 * sv + x1 * cv;
      u32 pk = (u32)f2bf(o0) | ((u32)f2bf(o1) << 16);
      *(u32*)&out[(size_t)grow * N + gn] = pk;
    }
  }
}

// ---------------------------------------------------------------------------
// Flash attention: causal, GQA (1 KV head). 64-row q block per workgroup,
// 4 waves x 16 q rows; 64-key blocks, double-buffered K/V in LDS.
// Q and K tiles staged by the Tensor Data Mover; V staged manually
// (transposed). HBM prefetch of block+1, fp32 online softmax.
// ---------------------------------------------------------------------------
__global__ __launch_bounds__(128) void flash_attn_kernel(
    const u16* __restrict__ qb, const u16* __restrict__ kb,
    const u16* __restrict__ vb, u16* __restrict__ ob)
{
  __shared__ u32 Qs32[64 * 65];        // Q block [qrow][hd] (dword pairs, pad 65)
  __shared__ u32 Ks32[2][64 * 65];     // K block [key][hd]
  __shared__ u16 Vt[2][128 * 66];      // V block transposed [hd][key] pad 66
  __shared__ u16 Ps[4 * 16 * 66];      // per-wave P strip (16x64) pad 66

  const int nqb  = T_ / 64;
  const int bid  = blockIdx.x;
  const int qblk = bid % nqb;
  const int h    = (bid / nqb) % H_;
  const int b    = bid / (nqb * H_);
  const int qbase = qblk * 64;

  const int tid  = threadIdx.x;
  const int wid  = tid >> 5;
  const int lane = tid & 31;
  const int l16  = lane & 15;
  const int lh   = lane >> 4;
  const int kg   = lh * 8;
  const int moff = lh * 8;

  // stage Q block once: [64 rows][128 hd] slice for this head
#if USE_TDM
  if (tid < 32)   // 64 rows x 64 dw, row stride 448 dw, +1 dw pad per 64 dw
    tdm_load_2d(&qb[(size_t)(b * T_ + qbase) * (H_ * HD_) + h * HD_],
                (u32)(size_t)&Qs32[0], 64u, 64u, (u32)((H_ * HD_) / 2), 5u);
#else
#pragma unroll 4
  for (int i = 0; i < 32; ++i) {
    int e = tid + i * 128;
    int row = e >> 6, c = e & 63;
    Qs32[row * 65 + c] =
        *(const u32*)&qb[(size_t)(b * T_ + qbase + row) * (H_ * HD_) + h * HD_ + 2 * c];
  }
#endif

  auto loadKV = [&](int kb2, int buf) {
    const int kbase = kb2 * 64;
#if USE_TDM
    if (tid < 32)   // wave 0 issues the K-tile DMA (uniform descriptor)
      tdm_load_2d(&kb[(size_t)(b * T_ + kbase) * HD_], (u32)(size_t)&Ks32[buf][0],
                  64u, 64u, 64u, 5u);
#endif
#pragma unroll 4
    for (int i = 0; i < 32; ++i) {
      int e = tid + i * 128;
      int row = e >> 6, c = e & 63;
      size_t g = (size_t)(b * T_ + kbase + row) * HD_ + 2 * c;
#if !USE_TDM
      Ks32[buf][row * 65 + c] = *(const u32*)&kb[g];
#endif
      u32 vvv = *(const u32*)&vb[g];
      Vt[buf][(2 * c) * 66 + row]     = (u16)(vvv & 0xffffu);
      Vt[buf][(2 * c + 1) * 66 + row] = (u16)(vvv >> 16);
    }
    if (kb2 + 1 < nqb) {               // prefetch block after this one
      size_t g2 = (size_t)(b * T_ + kbase + 64 + (tid >> 1)) * HD_ + (tid & 1) * 64;
      __builtin_prefetch(&kb[g2], 0, 0);
      __builtin_prefetch(&vb[g2], 0, 0);
    }
  };

  float m_run[8], l_run[8];
  FragF o[8];
#pragma unroll
  for (int r = 0; r < 8; ++r) { m_run[r] = -__builtin_inff(); l_run[r] = 0.f; }
#pragma unroll
  for (int ot = 0; ot < 8; ++ot)
#pragma unroll
    for (int r = 0; r < 8; ++r) o[ot].f[r] = 0.f;

  const float scale = 0.08838834764831845f;   // 1/sqrt(128)
  u16* myP = &Ps[wid * 16 * 66];
  const int qm = wid * 16 + l16;              // this lane's A-frag q row (local)

  loadKV(0, 0);
  int cur = 0;
  for (int kb2 = 0; kb2 <= qblk; ++kb2) {
#if USE_TDM
    if (tid < 32) __builtin_amdgcn_s_wait_tensorcnt(0);   // Q/K DMA done before barrier
#endif
    __syncthreads();                           // buf[cur] (and Qs) ready
    if (kb2 < qblk) loadKV(kb2 + 1, cur ^ 1);  // overlap next K/V with compute

    // ---- S = Q @ K^T (16x64 per wave), pipelined fragments ----
    FragF s[4];
#pragma unroll
    for (int nt = 0; nt < 4; ++nt)
#pragma unroll
      for (int r = 0; r < 8; ++r) s[nt].f[r] = 0.f;

    auto loadQA = [&](int c, FragB& qa) {
#pragma unroll
      for (int j = 0; j < 8; ++j)
        qa.u[j] = Qs32[qm * 65 + ((c * 32 + a_k0(kg, j)) >> 1)];
    };
    auto loadBK = [&](int c, int nt, FragB& bk) {
      const int n = nt * 16 + l16;
#pragma unroll
      for (int j = 0; j < 8; ++j)
        bk.u[j] = Ks32[cur][n * 65 + ((c * 32 + a_k0(kg, j)) >> 1)];
    };
    {
      FragB qa[2], bk[2];
      loadQA(0, qa[0]);
      loadBK(0, 0, bk[0]);
#pragma unroll
      for (int it = 0; it < 16; ++it) {        // it = c*4 + nt
        const int c = it >> 2, nt = it & 3;
        if (it + 1 < 16) loadBK((it + 1) >> 2, (it + 1) & 3, bk[(it + 1) & 1]);
        if (nt == 0 && c < 3) loadQA(c + 1, qa[(c + 1) & 1]);
        s[nt].v = wmma_bf16(qa[c & 1].v, bk[it & 1].v, s[nt].v);
      }
    }

    // ---- scale + causal mask (diagonal block only) ----
    const bool diag = (kb2 == qblk);
    float mt[8];
#pragma unroll
    for (int r = 0; r < 8; ++r) mt[r] = -__builtin_inff();
#pragma unroll
    for (int nt = 0; nt < 4; ++nt)
#pragma unroll
      for (int r = 0; r < 8; ++r) {
        float sv = s[nt].f[r] * scale;
        if (diag && (nt * 16 + l16 > wid * 16 + r + moff)) sv = -__builtin_inff();
        s[nt].f[r] = sv;
        mt[r] = fmaxf(mt[r], sv);
      }
#pragma unroll
    for (int r = 0; r < 8; ++r)
#pragma unroll
      for (int d = 1; d < 16; d <<= 1)
        mt[r] = fmaxf(mt[r], __shfl_xor(mt[r], d, 32));

    float alpha[8];
#pragma unroll
    for (int r = 0; r < 8; ++r) {
      float mn = fmaxf(m_run[r], mt[r]);
      alpha[r] = __expf(m_run[r] - mn);
      m_run[r] = mn;
    }
    float lt[8];
#pragma unroll
    for (int r = 0; r < 8; ++r) lt[r] = 0.f;
#pragma unroll
    for (int nt = 0; nt < 4; ++nt)
#pragma unroll
      for (int r = 0; r < 8; ++r) {
        float p = __expf(s[nt].f[r] - m_run[r]);
        s[nt].f[r] = p;
        lt[r] += p;
      }
#pragma unroll
    for (int r = 0; r < 8; ++r) {
#pragma unroll
      for (int d = 1; d < 16; d <<= 1)
        lt[r] += __shfl_xor(lt[r], d, 32);
      l_run[r] = l_run[r] * alpha[r] + lt[r];
    }
#pragma unroll
    for (int ot = 0; ot < 8; ++ot)
#pragma unroll
      for (int r = 0; r < 8; ++r) o[ot].f[r] *= alpha[r];

    // ---- stage P (C-layout -> LDS -> A-layout), private per-wave strip ----
#pragma unroll
    for (int nt = 0; nt < 4; ++nt)
#pragma unroll
      for (int r = 0; r < 8; ++r)
        myP[(r + moff) * 66 + nt * 16 + l16] = f2bf(s[nt].f[r]);

    // ---- O += P @ V (16x128 per wave), pipelined fragments ----
    auto loadPA = [&](int c, FragB& pa) {
#pragma unroll
      for (int j = 0; j < 8; ++j)
        pa.u[j] = *(const u32*)&myP[l16 * 66 + c * 32 + a_k0(kg, j)];
    };
    auto loadBV = [&](int c, int ot, FragB& bv) {
#pragma unroll
      for (int j = 0; j < 8; ++j)
        bv.u[j] = *(const u32*)&Vt[cur][(ot * 16 + l16) * 66 + c * 32 + a_k0(kg, j)];
    };
    {
      FragB pa[2], bv[2];
      loadPA(0, pa[0]);
      loadBV(0, 0, bv[0]);
#pragma unroll
      for (int it = 0; it < 16; ++it) {        // it = c*8 + ot
        const int c = it >> 3, ot = it & 7;
        if (it + 1 < 16) loadBV((it + 1) >> 3, (it + 1) & 7, bv[(it + 1) & 1]);
        if (it == 0) loadPA(1, pa[1]);
        o[ot].v = wmma_bf16(pa[c].v, bv[it & 1].v, o[ot].v);
      }
    }
    cur ^= 1;
  }

  float inv[8];
#pragma unroll
  for (int r = 0; r < 8; ++r) inv[r] = 1.0f / l_run[r];
  u16* optr = ob + (size_t)(b * T_ + qbase + wid * 16) * (H_ * HD_) + h * HD_;
#pragma unroll
  for (int ot = 0; ot < 8; ++ot)
#pragma unroll
    for (int r = 0; r < 8; ++r)
      optr[(size_t)(r + moff) * (H_ * HD_) + ot * 16 + l16] = f2bf(o[ot].f[r] * inv[r]);
}

// ---------------------------------------------------------------------------
extern "C" void kernel_launch(void* const* d_in, const int* in_sizes, int n_in,
                              void* d_out, int out_size, void* d_ws, size_t ws_size,
                              hipStream_t stream) {
  const float* x    = (const float*)d_in[0];
  const float* Wq   = (const float*)d_in[1];
  const float* Wk   = (const float*)d_in[2];
  const float* Wv   = (const float*)d_in[3];
  const float* Wo   = (const float*)d_in[4];
  const float* cosT = (const float*)d_in[5];
  const float* sinT = (const float*)d_in[6];

  const int M = B_ * T_;   // 8192
  u16* qbf = (u16*)d_ws;                        // M x 896 bf16 (RoPE'd)
  u16* kbf = qbf + (size_t)M * (H_ * HD_);      // M x 128 bf16 (RoPE'd)
  u16* vbf = kbf + (size_t)M * HD_;             // M x 128 bf16
  u16* abf = vbf + (size_t)M * HD_;             // M x 896 bf16 (attn out)

  dim3 blk(128);
  // Q/K projections with fused RoPE, V projection plain bf16
  gemm_wmma_kernel<true, 2><<<dim3((H_ * HD_) / 64, M / 128), blk, 0, stream>>>(
      x, Wq, qbf, M, H_ * HD_, D_, cosT, sinT);
  gemm_wmma_kernel<true, 2><<<dim3(HD_ / 64, M / 128), blk, 0, stream>>>(
      x, Wk, kbf, M, HD_, D_, cosT, sinT);
  gemm_wmma_kernel<true, 1><<<dim3(HD_ / 64, M / 128), blk, 0, stream>>>(
      x, Wv, vbf, M, HD_, D_, nullptr, nullptr);

  // causal flash attention (TDM-staged Q/K tiles)
  flash_attn_kernel<<<dim3(B_ * H_ * (T_ / 64)), blk, 0, stream>>>(qbf, kbf, vbf, abf);

  // output projection -> fp32 (TDM-staged A tiles)
  gemm_wmma_kernel<false, 0><<<dim3(D_ / 64, M / 128), blk, 0, stream>>>(
      abf, Wo, (float*)d_out, M, D_, H_ * HD_, nullptr, nullptr);
}
